// OutlookAttentionLayer_82437602279999
// MI455X (gfx1250) — matrix operand order repaired and checked by reading, not compiled, not run on hardware
//
#include <hip/hip_runtime.h>
#include <hip/hip_bf16.h>
#include <cstdint>

// ---------------------------------------------------------------- types
typedef float          v8f   __attribute__((ext_vector_type(8)));
typedef __bf16         v16bf __attribute__((ext_vector_type(16)));
typedef unsigned int   u32x4 __attribute__((ext_vector_type(4)));
typedef unsigned int   u32x8 __attribute__((ext_vector_type(8)));

#define HEADS 8
#define C_DIM 384
#define HD    48
#define K2    9
#define ACH   648           // K^4 * HEADS
#define NPOS  32768         // 8 * 64 * 64
#define HW    4096
#define SCALE 0.14433756729740643f   // 48^-0.5
#define EPS   1e-6f

// LDS X-tile: 256 rows x 32 k of bf16, row stride padded to 80B (40 u16)
// -> b128 reads start at banks {20*r mod 64}: conflict-free across a half-wave.
#define XT_STRIDE_U16 40

// ---------------------------------------------------------------- helpers
__device__ __forceinline__ unsigned short f2bf(float f) {
    unsigned int u = __builtin_bit_cast(unsigned int, f);
    u += 0x7FFFu + ((u >> 16) & 1u);           // round-to-nearest-even
    return (unsigned short)(u >> 16);
}
__device__ __forceinline__ float bf2f(unsigned short h) {
    return __builtin_bit_cast(float, (unsigned int)h << 16);
}

// two contiguous 16B chunks -> one 16-element bf16 fragment
__device__ __forceinline__ v16bf load_frag2(const unsigned short* p0,
                                            const unsigned short* p1) {
    u32x4 a = *(const u32x4*)p0;
    u32x4 b = *(const u32x4*)p1;
    u32x8 r;
    r[0] = a[0]; r[1] = a[1]; r[2] = a[2]; r[3] = a[3];
    r[4] = b[0]; r[5] = b[1]; r[6] = b[2]; r[7] = b[3];
    return __builtin_bit_cast(v16bf, r);
}

__device__ __forceinline__ v8f wmma_bf16(v16bf a, v16bf b, v8f c) {
    return __builtin_amdgcn_wmma_f32_16x16x32_bf16(
        /*neg_a=*/false, a, /*neg_b=*/false, b,
        /*c_mod=*/(short)0, c, /*reuse_a=*/false, /*reuse_b=*/false);
}

// async copy of one 16B chunk global -> LDS (tracked by ASYNCcnt)
__device__ __forceinline__ void async_g2l_b128(unsigned short* lds_dst,
                                               const unsigned short* gsrc) {
    unsigned laddr = (unsigned)(uintptr_t)lds_dst;           // addr[31:0] = LDS offset
    unsigned long long gaddr = (unsigned long long)(uintptr_t)gsrc;
    asm volatile("global_load_async_to_lds_b128 %0, %1, off"
                 :: "v"(laddr), "v"(gaddr) : "memory");
}
__device__ __forceinline__ void wait_async0() {
    asm volatile("s_wait_asynccnt 0x0" ::: "memory");
}

// ---------------------------------------------------------------- GEMM core
// Block tile: M=64 (2 waves) x N=256 (4 waves); wave tile 32x64 (2x4 of 16x16).
// A = W[M x 384] row-major bf16 (global, L2-resident);
// B = X[N x 384] row-major bf16, k-slice staged in LDS via async-to-LDS.
__device__ __forceinline__ void gemm_block(const unsigned short* __restrict__ W,
                                           int mclamp, int mblk,
                                           const unsigned short* __restrict__ X,
                                           int pbase,
                                           unsigned short* ldsx,
                                           v8f acc[2][4]) {
    const int tid  = threadIdx.x;
    const int lane = tid & 31;
    const int wave = tid >> 5;
    const int wm   = wave & 1;
    const int wn   = wave >> 1;
    const int l16  = lane & 15;
    const int hi   = lane >> 4;
    const int cchunk = tid & 3;         // 4 threads cover one 64B row slice
    const int rbase  = tid >> 2;

#pragma unroll 1
    for (int k0 = 0; k0 < C_DIM; k0 += 32) {
        // ---- stage X[pbase..pbase+255, k0..k0+31] into LDS (async DMA path)
#pragma unroll
        for (int r = 0; r < 4; ++r) {
            const int row = rbase + r * 64;
            async_g2l_b128(ldsx + row * XT_STRIDE_U16 + cchunk * 8,
                           X + (size_t)(pbase + row) * C_DIM + k0 + cchunk * 8);
        }
        wait_async0();
        __syncthreads();

        // ---- A fragments (direct global)
        v16bf aF[2];
#pragma unroll
        for (int mt = 0; mt < 2; ++mt) {
            int r = mblk + wm * 32 + mt * 16 + l16;
            if (r >= mclamp) r = mclamp - 1;
            const unsigned short* base = W + (size_t)r * C_DIM + k0 + hi * 8;
            aF[mt] = load_frag2(base, base + 16);
        }
        // ---- B fragments (from LDS)
        v16bf bF[4];
#pragma unroll
        for (int nt = 0; nt < 4; ++nt) {
            const int prow = wn * 64 + nt * 16 + l16;
            const unsigned short* base = ldsx + prow * XT_STRIDE_U16 + hi * 8;
            bF[nt] = load_frag2(base, base + 16);
        }
#pragma unroll
        for (int mt = 0; mt < 2; ++mt)
#pragma unroll
            for (int nt = 0; nt < 4; ++nt)
                acc[mt][nt] = wmma_bf16(aF[mt], bF[nt], acc[mt][nt]);

        __syncthreads();                 // protect LDS before next stage
    }
}

// ---------------------------------------------------------------- kernels
__global__ __launch_bounds__(256) void cvt_bf16_kernel(const float* __restrict__ src,
                                                       unsigned short* __restrict__ dst,
                                                       int n) {
    int i = blockIdx.x * 256 + threadIdx.x;
    if (i < n) dst[i] = f2bf(src[i]);
}

// LayerNorm over C, output bf16 [pos, C] row-major (LDS transpose for coalesced stores)
__global__ __launch_bounds__(256) void ln_kernel(const float* __restrict__ feats,
                                                 const float* __restrict__ gamma,
                                                 const float* __restrict__ beta,
                                                 unsigned short* __restrict__ X) {
    __shared__ unsigned int lds[256 * 16];          // 16 KB
    const int tid  = threadIdx.x;
    const int pos0 = blockIdx.x * 256;
    const int pos  = pos0 + tid;
    const int b    = pos >> 12;
    const int hw   = pos & (HW - 1);
    const size_t base = (size_t)b * C_DIM * HW + hw;

    float s = 0.f, ss = 0.f;
#pragma unroll 4
    for (int c = 0; c < C_DIM; ++c) {
        float f = feats[base + (size_t)c * HW];
        s += f; ss += f * f;
    }
    const float mean = s * (1.f / C_DIM);
    const float var  = ss * (1.f / C_DIM) - mean * mean;
    const float inv  = rsqrtf(var + EPS);

    unsigned int* __restrict__ xo = (unsigned int*)X;
    for (int c0 = 0; c0 < C_DIM; c0 += 32) {
#pragma unroll
        for (int cd = 0; cd < 16; ++cd) {
            const int c = c0 + 2 * cd;
            float f0 = (feats[base + (size_t)c * HW]       - mean) * inv * gamma[c]     + beta[c];
            float f1 = (feats[base + (size_t)(c + 1) * HW] - mean) * inv * gamma[c + 1] + beta[c + 1];
            lds[tid * 16 + cd] = (unsigned int)f2bf(f0) | ((unsigned int)f2bf(f1) << 16);
        }
        __syncthreads();
#pragma unroll
        for (int p = 0; p < 16; ++p) {
            const int g  = p * 256 + tid;
            const int r  = g >> 4;
            const int cd = g & 15;
            xo[(size_t)(pos0 + r) * (C_DIM / 2) + (c0 >> 1) + cd] = lds[g];
        }
        __syncthreads();
    }
}

// v = x * w_v^T + b_v  -> bf16 [pos, C]
__global__ __launch_bounds__(256) void gemm_v_kernel(const unsigned short* __restrict__ W,
                                                     const unsigned short* __restrict__ X,
                                                     const float* __restrict__ bias,
                                                     unsigned short* __restrict__ Vout) {
    __shared__ unsigned short ldsx[256 * XT_STRIDE_U16];
    const int mblk  = blockIdx.x * 64;
    const int pbase = blockIdx.y * 256;
    v8f acc[2][4] = {};
    gemm_block(W, C_DIM, mblk, X, pbase, ldsx, acc);

    const int lane = threadIdx.x & 31;
    const int wave = threadIdx.x >> 5;
    const int n16  = lane & 15;
    const int hi   = lane >> 4;
#pragma unroll
    for (int mt = 0; mt < 2; ++mt)
#pragma unroll
        for (int nt = 0; nt < 4; ++nt) {
            const int pos = pbase + (wave >> 1) * 64 + nt * 16 + n16;
            const int oB  = mblk + (wave & 1) * 32 + mt * 16 + hi * 8;
            unsigned int pk[4] = {0u, 0u, 0u, 0u};
#pragma unroll
            for (int vi = 0; vi < 8; ++vi) {
                float f = acc[mt][nt][vi] + bias[oB + vi];
                pk[vi >> 1] |= (unsigned int)f2bf(f) << ((vi & 1) * 16);
            }
            u32x4 st; st[0] = pk[0]; st[1] = pk[1]; st[2] = pk[2]; st[3] = pk[3];
            *(u32x4*)(Vout + (size_t)pos * C_DIM + oB) = st;
        }
}

// a = x * w_attn^T + b_attn -> fp32 [pos, 648]
__global__ __launch_bounds__(256) void gemm_attn_kernel(const unsigned short* __restrict__ W,
                                                        const unsigned short* __restrict__ X,
                                                        const float* __restrict__ bias,
                                                        float* __restrict__ A) {
    __shared__ unsigned short ldsx[256 * XT_STRIDE_U16];
    const int mblk  = blockIdx.x * 64;
    const int pbase = blockIdx.y * 256;
    v8f acc[2][4] = {};
    gemm_block(W, ACH, mblk, X, pbase, ldsx, acc);

    const int lane = threadIdx.x & 31;
    const int wave = threadIdx.x >> 5;
    const int n16  = lane & 15;
    const int hi   = lane >> 4;
#pragma unroll
    for (int mt = 0; mt < 2; ++mt)
#pragma unroll
        for (int nt = 0; nt < 4; ++nt) {
            const int pos = pbase + (wave >> 1) * 64 + nt * 16 + n16;
            const int oB  = mblk + (wave & 1) * 32 + mt * 16 + hi * 8;
            float* __restrict__ row = A + (size_t)pos * ACH;
#pragma unroll
            for (int vi = 0; vi < 8; ++vi) {
                const int o = oB + vi;
                if (o < ACH) row[o] = acc[mt][nt][vi] + bias[o];
            }
        }
}

// sim = a * SCALE (written to d_out), then softmax over last 9 written back into A
__global__ __launch_bounds__(256) void softmax_kernel(float* __restrict__ A,
                                                      float* __restrict__ sim) {
    const int idx = blockIdx.x * 256 + threadIdx.x;
    if (idx >= NPOS * HEADS * K2) return;
    const int pos  = idx / (HEADS * K2);
    const int rem  = idx - pos * (HEADS * K2);
    const int head = rem / K2;
    const int kap  = rem - head * K2;

    float* __restrict__ row = A + (size_t)pos * ACH + head * 81 + kap * 9;
    float v[9], mx = -3.4e38f;
#pragma unroll
    for (int j = 0; j < 9; ++j) { v[j] = row[j] * SCALE; mx = fmaxf(mx, v[j]); }
    float sum = 0.f, e[9];
#pragma unroll
    for (int j = 0; j < 9; ++j) { e[j] = __expf(v[j] - mx); sum += e[j]; }
    const float rs = 1.f / sum;

    const int b  = pos >> 12;
    const int hw = pos & (HW - 1);
    float* __restrict__ so = sim + ((size_t)((b * HEADS + head) * HW + hw)) * 81 + kap * 9;
#pragma unroll
    for (int j = 0; j < 9; ++j) { so[j] = v[j]; row[j] = e[j] * rs; }
}

// folded[q,c] = sum_{k,k'} attn[n(k), k, k'] * v[(qy+dy, qx+dx), c]  (fused apply+fold)
__global__ __launch_bounds__(384) void apply_fold_kernel(const float* __restrict__ Attn,
                                                         const unsigned short* __restrict__ V,
                                                         unsigned short* __restrict__ Folded) {
    __shared__ float sa[HEADS * 81];
    const int q  = blockIdx.x;
    const int b  = q >> 12;
    const int hw = q & (HW - 1);
    const int qy = hw >> 6;
    const int qx = hw & 63;
    const int c  = threadIdx.x;
    const int head = c / HD;

    for (int t = threadIdx.x; t < HEADS * 81; t += 384) {
        const int ht  = t / 81;
        const int r   = t - ht * 81;
        const int kap = r / 9;             // fold index (i,j)
        const int kp  = r - kap * 9;       // unfold index (i',j')
        const int i   = kap / 3, j = kap - i * 3;
        const int ny  = qy + 1 - i, nx = qx + 1 - j;
        float val = 0.f;
        if ((unsigned)ny < 64u && (unsigned)nx < 64u)
            val = Attn[((size_t)(b * HW + ny * 64 + nx)) * ACH + ht * 81 + kap * 9 + kp];
        sa[t] = val;
    }
    __syncthreads();

    float vv[25];
#pragma unroll
    for (int dy = 0; dy < 5; ++dy)
#pragma unroll
        for (int dx = 0; dx < 5; ++dx) {
            const int yy = qy + dy - 2, xx = qx + dx - 2;
            float val = 0.f;
            if ((unsigned)yy < 64u && (unsigned)xx < 64u)
                val = bf2f(V[((size_t)(b * HW + yy * 64 + xx)) * C_DIM + c]);
            vv[dy * 5 + dx] = val;
        }

    float acc = 0.f;
    const float* __restrict__ ah = sa + head * 81;
#pragma unroll
    for (int kap = 0; kap < 9; ++kap) {
        const int i = kap / 3, j = kap - i * 3;
#pragma unroll
        for (int kp = 0; kp < 9; ++kp) {
            const int ip = kp / 3, jp = kp - ip * 3;
            acc += ah[kap * 9 + kp] * vv[(ip - i + 2) * 5 + (jp - j + 2)];
        }
    }
    Folded[(size_t)q * C_DIM + c] = f2bf(acc);
}

// y = folded * w_out^T + b_out -> fp32 channel-major (b, C, h, w)
__global__ __launch_bounds__(256) void gemm_out_kernel(const unsigned short* __restrict__ W,
                                                       const unsigned short* __restrict__ X,
                                                       const float* __restrict__ bias,
                                                       float* __restrict__ Y) {
    __shared__ unsigned short ldsx[256 * XT_STRIDE_U16];
    const int mblk  = blockIdx.x * 64;
    const int pbase = blockIdx.y * 256;
    v8f acc[2][4] = {};
    gemm_block(W, C_DIM, mblk, X, pbase, ldsx, acc);

    const int lane = threadIdx.x & 31;
    const int wave = threadIdx.x >> 5;
    const int n16  = lane & 15;
    const int hi   = lane >> 4;
#pragma unroll
    for (int mt = 0; mt < 2; ++mt)
#pragma unroll
        for (int nt = 0; nt < 4; ++nt) {
            const int pos = pbase + (wave >> 1) * 64 + nt * 16 + n16;
            const int b   = pos >> 12;
            const int hw  = pos & (HW - 1);
            const int oB  = mblk + (wave & 1) * 32 + mt * 16 + hi * 8;
#pragma unroll
            for (int vi = 0; vi < 8; ++vi) {
                const int o = oB + vi;
                Y[((size_t)(b * C_DIM + o)) * HW + hw] = acc[mt][nt][vi] + bias[o];
            }
        }
}

// ---------------------------------------------------------------- launch
extern "C" void kernel_launch(void* const* d_in, const int* in_sizes, int n_in,
                              void* d_out, int out_size, void* d_ws, size_t ws_size,
                              hipStream_t stream) {
    const float* feats  = (const float*)d_in[0];
    const float* gamma  = (const float*)d_in[1];
    const float* beta   = (const float*)d_in[2];
    const float* w_v    = (const float*)d_in[3];
    const float* b_v    = (const float*)d_in[4];
    const float* w_attn = (const float*)d_in[5];
    const float* b_attn = (const float*)d_in[6];
    const float* w_out  = (const float*)d_in[7];
    const float* b_out  = (const float*)d_in[8];

    float* y   = (float*)d_out;                        // 8*384*4096 floats
    float* sim = y + (size_t)8 * C_DIM * HW;           // 8*8*4096*81 floats

    char* ws = (char*)d_ws;
    unsigned short* Xb  = (unsigned short*)(ws);                    // bf16 x; later folded
    unsigned short* Vb  = (unsigned short*)(ws + 25165824);
    float*          Ab  = (float*)(ws + 50331648);                  // a / attn in-place
    unsigned short* Wvb = (unsigned short*)(ws + 135266304);
    unsigned short* Wab = (unsigned short*)(ws + 135561216);
    unsigned short* Wob = (unsigned short*)(ws + 136058880);

    // weight casts
    cvt_bf16_kernel<<<(C_DIM * C_DIM + 255) / 256, 256, 0, stream>>>(w_v, Wvb, C_DIM * C_DIM);
    cvt_bf16_kernel<<<(ACH * C_DIM + 255) / 256, 256, 0, stream>>>(w_attn, Wab, ACH * C_DIM);
    cvt_bf16_kernel<<<(C_DIM * C_DIM + 255) / 256, 256, 0, stream>>>(w_out, Wob, C_DIM * C_DIM);

    // layernorm -> bf16 [pos, C]
    ln_kernel<<<NPOS / 256, 256, 0, stream>>>(feats, gamma, beta, Xb);

    // WMMA GEMMs (block tile 64 x 256, async-to-LDS staged X)
    gemm_v_kernel<<<dim3(C_DIM / 64, NPOS / 256), 256, 0, stream>>>(Wvb, Xb, b_v, Vb);
    gemm_attn_kernel<<<dim3((ACH + 63) / 64, NPOS / 256), 256, 0, stream>>>(Wab, Xb, b_attn, Ab);

    // sim out + softmax (in place on Ab)
    softmax_kernel<<<(NPOS * HEADS * K2) / 256, 256, 0, stream>>>(Ab, sim);

    // fused attention-apply + fold; folded aliases Xb (x no longer needed)
    apply_fold_kernel<<<NPOS, 384, 0, stream>>>(Ab, Vb, Xb);

    // final projection into channel-major y
    gemm_out_kernel<<<dim3(C_DIM / 64, NPOS / 256), 256, 0, stream>>>(Wob, Xb, b_out, y);
}
